// QBBLinear_76562087019018
// MI455X (gfx1250) — matrix operand order repaired and checked by reading, not compiled, not run on hardware
//
#include <hip/hip_runtime.h>

// Problem constants (match reference)
#define IN_DIM   4096
#define OUT_DIM  4096
#define M_ROWS   4096   // B*S = 2*2048
#define KBASES   4

// GEMM tiling: block tile 128(M) x 256(N), 8 waves, each wave 64x64 (16 WMMA)
#define BM 128
#define BN 256
#define BK 32
#define LDK 40                    // padded LDS row stride (bf16): 80B = 5*16B, conflict-free
#define PLANE_A (BM * LDK)        // 5120 elems
#define PLANE_W (BN * LDK)        // 10240 elems
#define BUF     (PLANE_A + PLANE_W)   // 15360 elems per double-buffer slot
#define BUFB    (BUF * 2)             // bytes per slot (30720)

typedef __attribute__((ext_vector_type(16))) __bf16         v16bf;
typedef __attribute__((ext_vector_type(8)))  float          v8f;
typedef __attribute__((ext_vector_type(4)))  unsigned int   u32x4;
typedef __attribute__((ext_vector_type(4)))  float          f32x4;
typedef __attribute__((ext_vector_type(8)))  unsigned short u16x8;
typedef __attribute__((ext_vector_type(8)))  signed char    i8x8;

union Frag { u32x4 q[2]; v16bf v; };

static __device__ __forceinline__ unsigned short f32_to_bf16_rne(float f) {
    unsigned int u = __float_as_uint(f);
    unsigned int r = 0x7FFFu + ((u >> 16) & 1u);
    return (unsigned short)((u + r) >> 16);
}

// ---- CDNA5 async global->LDS (ASYNCcnt-tracked); offset: adds to BOTH addresses ----
static __device__ __forceinline__ void async_ld_b128(unsigned lds_off, const void* g) {
    asm volatile("global_load_async_to_lds_b128 %0, %1, off"
                 :: "v"(lds_off), "v"(g) : "memory");
}
static __device__ __forceinline__ void async_ld_b128_o16(unsigned lds_off, const void* g) {
    asm volatile("global_load_async_to_lds_b128 %0, %1, off offset:16"
                 :: "v"(lds_off), "v"(g) : "memory");
}
static __device__ __forceinline__ void async_ld_b128_o32(unsigned lds_off, const void* g) {
    asm volatile("global_load_async_to_lds_b128 %0, %1, off offset:32"
                 :: "v"(lds_off), "v"(g) : "memory");
}
static __device__ __forceinline__ void async_ld_b128_o48(unsigned lds_off, const void* g) {
    asm volatile("global_load_async_to_lds_b128 %0, %1, off offset:48"
                 :: "v"(lds_off), "v"(g) : "memory");
}
static __device__ __forceinline__ void wait_async0() {
    asm volatile("s_wait_asynccnt 0x0" ::: "memory");
}

// ---------------- Kernel 1: x (f32) -> bf16 ----------------
__global__ void __launch_bounds__(256)
cvt_x_bf16(const float* __restrict__ x, unsigned short* __restrict__ x16) {
    int i = (blockIdx.x * 256 + threadIdx.x) * 8;
    f32x4 a = *(const f32x4*)(x + i);
    f32x4 b = *(const f32x4*)(x + i + 4);
    u16x8 o;
#pragma unroll
    for (int j = 0; j < 4; ++j) {
        o[j]     = f32_to_bf16_rne(a[j]);
        o[j + 4] = f32_to_bf16_rne(b[j]);
    }
    *(u16x8*)(x16 + i) = o;
}

// ------- Kernel 2: W_q = sum_k alpha_k * bases_k  (bf16) -------
__global__ void __launch_bounds__(256)
build_w_bf16(const signed char* __restrict__ bases,
             const float* __restrict__ alphas,
             unsigned short* __restrict__ w16) {
    const size_t KOI = (size_t)OUT_DIM * IN_DIM;
    size_t i = ((size_t)blockIdx.x * 256 + threadIdx.x) * 8;
    float a0 = alphas[0], a1 = alphas[1], a2 = alphas[2], a3 = alphas[3];
    i8x8 b0 = *(const i8x8*)(bases + 0 * KOI + i);
    i8x8 b1 = *(const i8x8*)(bases + 1 * KOI + i);
    i8x8 b2 = *(const i8x8*)(bases + 2 * KOI + i);
    i8x8 b3 = *(const i8x8*)(bases + 3 * KOI + i);
    u16x8 o;
#pragma unroll
    for (int j = 0; j < 8; ++j) {
        float w = a0 * (float)b0[j] + a1 * (float)b1[j]
                + a2 * (float)b2[j] + a3 * (float)b3[j];
        o[j] = f32_to_bf16_rne(w);
    }
    *(u16x8*)(w16 + i) = o;
}

// ---------------- Kernel 3: y = x @ W^T  (bf16 WMMA, f32 acc) ----------------
// A: x16 [M][K] row-major; W: w16 [N][K] row-major; out: f32 [M][N]
__global__ void __launch_bounds__(256)
qbb_gemm_bf16(const unsigned short* __restrict__ A,
              const unsigned short* __restrict__ W,
              float* __restrict__ out) {
    __shared__ unsigned short lds[2 * BUF];   // [buf][A plane | W plane], 60 KB

    const int tid   = threadIdx.x;
    const int lane  = tid & 31;
    const int lr    = lane & 15;
    const int lhalf = lane >> 4;            // 0 or 1
    const int wid   = tid >> 5;             // 8 waves
    const int wm    = (wid & 1) * 64;       // wave M offset in block tile
    const int wn    = (wid >> 1) * 64;      // wave N offset in block tile

    const int bm = blockIdx.y * BM;
    const int bn = blockIdx.x * BN;

    // --- async tile-fill mapping ---
    // A plane: thread covers 32B of a row (2 x b128): row tid/2, col (tid&1)*16 elems
    const int arow = tid >> 1;
    const int acol = (tid & 1) * 16;
    // W plane: thread covers a full 64B row (4 x b128): row tid
    const int wrow = tid;

    const unsigned short* gA = A + (size_t)(bm + arow) * IN_DIM + acol;
    const unsigned short* gW = W + (size_t)(bn + wrow) * IN_DIM;

    unsigned ldsA = (unsigned)(unsigned long long)&lds[arow * LDK + acol];
    unsigned ldsW = (unsigned)(unsigned long long)&lds[PLANE_A + wrow * LDK];

    // fragment-read base addresses (ISA 7.12.2 layouts); mt/nt offsets are immediates
    const int a_lk = lhalf * 8;             // A: two 16B K-chunks at a_lk, a_lk+16
    const int b_lk = lhalf * 8;             // B: two 16B K-chunks at b_lk, b_lk+16

    int rdA = (wm + lr) * LDK + a_lk;            // elems, buffer 0
    int rdW = PLANE_A + (wn + lr) * LDK + b_lk;  // elems, buffer 0
    int d   = BUF;

    v8f acc[4][4] = {};

    // prologue: async-fill buffer 0
    async_ld_b128     (ldsA, gA);
    async_ld_b128_o16 (ldsA, gA);
    async_ld_b128     (ldsW, gW);
    async_ld_b128_o16 (ldsW, gW);
    async_ld_b128_o32 (ldsW, gW);
    async_ld_b128_o48 (ldsW, gW);
    wait_async0();
    __syncthreads();

    // in-loop async writes go to buffer 1 first
    ldsA += BUFB;
    ldsW += BUFB;
    const unsigned short* gAp = gA + BK;
    const unsigned short* gWp = gW + BK;

    const int NK = IN_DIM / BK;             // 128 k-steps
#pragma clang loop unroll(disable)
    for (int ks = 0; ks < NK - 1; ++ks) {
        // kick off async fill of the other buffer (overlaps with WMMA below)
        async_ld_b128     (ldsA, gAp);
        async_ld_b128_o16 (ldsA, gAp);
        async_ld_b128     (ldsW, gWp);
        async_ld_b128_o16 (ldsW, gWp);
        async_ld_b128_o32 (ldsW, gWp);
        async_ld_b128_o48 (ldsW, gWp);
        gAp += BK;
        gWp += BK;

        // compute from current buffer
        Frag afr[4], bfr[4];
#pragma unroll
        for (int mt = 0; mt < 4; ++mt) {
            afr[mt].q[0] = *(const u32x4*)&lds[rdA + mt * (16 * LDK)];
            afr[mt].q[1] = *(const u32x4*)&lds[rdA + mt * (16 * LDK) + 16];
        }
#pragma unroll
        for (int nt = 0; nt < 4; ++nt) {
            bfr[nt].q[0] = *(const u32x4*)&lds[rdW + nt * (16 * LDK)];
            bfr[nt].q[1] = *(const u32x4*)&lds[rdW + nt * (16 * LDK) + 16];
        }
#pragma unroll
        for (int mt = 0; mt < 4; ++mt)
#pragma unroll
            for (int nt = 0; nt < 4; ++nt)
                acc[mt][nt] = __builtin_amdgcn_wmma_f32_16x16x32_bf16(
                    false, afr[mt].v, false, bfr[nt].v,
                    (short)0, acc[mt][nt], false, false);

        // flip buffers
        rdA += d;
        rdW += d;
        ldsA = (unsigned)((int)ldsA - 2 * d);
        ldsW = (unsigned)((int)ldsW - 2 * d);
        d = -d;

        // publish the async-filled buffer to the workgroup
        wait_async0();
        __syncthreads();
    }

    // final k-step: compute only
    {
        Frag afr[4], bfr[4];
#pragma unroll
        for (int mt = 0; mt < 4; ++mt) {
            afr[mt].q[0] = *(const u32x4*)&lds[rdA + mt * (16 * LDK)];
            afr[mt].q[1] = *(const u32x4*)&lds[rdA + mt * (16 * LDK) + 16];
        }
#pragma unroll
        for (int nt = 0; nt < 4; ++nt) {
            bfr[nt].q[0] = *(const u32x4*)&lds[rdW + nt * (16 * LDK)];
            bfr[nt].q[1] = *(const u32x4*)&lds[rdW + nt * (16 * LDK) + 16];
        }
#pragma unroll
        for (int mt = 0; mt < 4; ++mt)
#pragma unroll
            for (int nt = 0; nt < 4; ++nt)
                acc[mt][nt] = __builtin_amdgcn_wmma_f32_16x16x32_bf16(
                    false, afr[mt].v, false, bfr[nt].v,
                    (short)0, acc[mt][nt], false, false);
    }

    // epilogue: C/D layout -> VGPR r holds (M = r + 8*lhalf, N = lr)
    const int orow0 = bm + wm + lhalf * 8;
    const int ocol0 = bn + wn + lr;
#pragma unroll
    for (int mt = 0; mt < 4; ++mt) {
#pragma unroll
        for (int nt = 0; nt < 4; ++nt) {
            float* po = out + (size_t)(orow0 + mt * 16) * OUT_DIM + (ocol0 + nt * 16);
#pragma unroll
            for (int r = 0; r < 8; ++r)
                po[(size_t)r * OUT_DIM] = acc[mt][nt][r];
        }
    }
}

extern "C" void kernel_launch(void* const* d_in, const int* in_sizes, int n_in,
                              void* d_out, int out_size, void* d_ws, size_t ws_size,
                              hipStream_t stream) {
    (void)in_sizes; (void)n_in; (void)out_size; (void)ws_size;
    const float*       x      = (const float*)d_in[0];
    const float*       alphas = (const float*)d_in[1];
    const signed char* bases  = (const signed char*)d_in[2];
    float*             out    = (float*)d_out;

    unsigned short* x16 = (unsigned short*)d_ws;                       // 32 MB
    unsigned short* w16 = x16 + (size_t)M_ROWS * IN_DIM;               // 32 MB

    cvt_x_bf16<<<(M_ROWS * IN_DIM) / (256 * 8), 256, 0, stream>>>(x, x16);
    build_w_bf16<<<(OUT_DIM * IN_DIM) / (256 * 8), 256, 0, stream>>>(bases, alphas, w16);

    dim3 grid(OUT_DIM / BN, M_ROWS / BM);
    qbb_gemm_bf16<<<grid, 256, 0, stream>>>(x16, w16, out);
}
